// UFNO3D_24300924960930
// MI455X (gfx1250) — matrix operand order repaired and checked by reading, not compiled, not run on hardware
//
#include <hip/hip_runtime.h>
#include <math.h>

#define TPB 256

typedef __attribute__((ext_vector_type(16))) __bf16 v16bf;
typedef __attribute__((ext_vector_type(8)))  float  v8f;

__device__ __forceinline__ unsigned short f2bf(float f) {
  union { float f; unsigned u; } v; v.f = f;
  unsigned u = v.u;
  return (unsigned short)((u + 0x7FFFu + ((u >> 16) & 1u)) >> 16);
}

__device__ __forceinline__ unsigned pk2bf(float a, float b) {
  return (unsigned)f2bf(a) | ((unsigned)f2bf(b) << 16);
}

__device__ __forceinline__ float gelu_exact(float v) {
  return 0.5f * v * (1.0f + erff(v * 0.70710678118654752f));
}

// =====================================================================
// WMMA GEMM: out[b,m,n] = sum_k wgt[m,k] * act[b,k,n] (+bias[m]) (opt GELU)
// act: (B, K, N) fp32, channel stride == N.  wgt: (M,K) fp32 row-major.
// N multiple of 256 at every call site (S in {512,4096,32768,262144}).
// Block = 128 threads = 4 waves; tile = 32(M) x 256(N).
// Per K-step: stage 32x32 weights (2KB) + 32x256 activations (16KB) in LDS,
// each wave issues 8 x v_wmma_f32_16x16x32_bf16 (2 A-frags x 4 B-frags).
// =====================================================================
__global__ __launch_bounds__(128)
void k_gemm_wmma(const float* __restrict__ act, long actBS,
                 const float* __restrict__ wgt,
                 const float* __restrict__ bias,
                 float* __restrict__ out, long outBS,
                 int M, int K, int Kpad, int N, int fuseGelu)
{
  __shared__ __align__(16) unsigned short Wt[32 * 32];    //  2 KB, [m][k]
  __shared__ __align__(16) unsigned short At[256 * 32];   // 16 KB, [n][k]

  const int  tid  = threadIdx.x;
  const int  wv   = tid >> 5;
  const int  lane = tid & 31;
  const long n0   = (long)blockIdx.x * 256;
  const int  m0   = blockIdx.y * 32;
  const float* A  = act + (long)blockIdx.z * actBS;
  float*       O  = out + (long)blockIdx.z * outBS;

  v8f acc[2][4];
#pragma unroll
  for (int mt = 0; mt < 2; ++mt)
#pragma unroll
    for (int t = 0; t < 4; ++t)
#pragma unroll
      for (int r = 0; r < 8; ++r) acc[mt][t][r] = 0.0f;

  const int row  = lane & 15;
  const int half = lane >> 4;
  const int wn   = wv * 64;

  for (int k0 = 0; k0 < Kpad; k0 += 32) {
    const bool fullK = (k0 + 32 <= K);

    // ---- stage 32x32 weight tile -> Wt[m][k] (bf16, zero padded) ----
    if (fullK && (m0 + 32 <= M)) {
#pragma unroll
      for (int j = 0; j < 4; ++j) {
        int idx = tid + j * 128;          // 0..511 bf16-pairs
        int r   = idx >> 4;               // 0..31
        int kp  = (idx & 15) * 2;         // 0,2,..,30
        const float* wp = wgt + (long)(m0 + r) * K + k0 + kp;
        *(unsigned*)&Wt[r * 32 + kp] = pk2bf(wp[0], wp[1]);
      }
    } else {
#pragma unroll
      for (int j = 0; j < 4; ++j) {
        int idx = tid + j * 128;
        int r   = idx >> 4;
        int kp  = (idx & 15) * 2;
        int m   = m0 + r;
        const float* wp = wgt + (long)m * K + k0 + kp;
        float w0 = (m < M && k0 + kp + 0 < K) ? wp[0] : 0.0f;
        float w1 = (m < M && k0 + kp + 1 < K) ? wp[1] : 0.0f;
        *(unsigned*)&Wt[r * 32 + kp] = pk2bf(w0, w1);
      }
    }

    // ---- stage 32x256 activation tile -> At[n][k] (bf16) ----
    // thread handles nn = tid and nn = tid+128; kg walks 0..7 via row ptr.
    const float* pr = A + (long)k0 * N + n0 + tid;
    const long  N1 = N;
    if (fullK) {
#pragma unroll
      for (int jj = 0; jj < 8; ++jj) {
        float v0 = pr[0],       v1 = pr[N1],       v2 = pr[2 * N1],       v3 = pr[3 * N1];
        float u0 = pr[128],     u1 = pr[N1 + 128], u2 = pr[2 * N1 + 128], u3 = pr[3 * N1 + 128];
        *(uint2*)&At[tid * 32 + jj * 4]         = make_uint2(pk2bf(v0, v1), pk2bf(v2, v3));
        *(uint2*)&At[(tid + 128) * 32 + jj * 4] = make_uint2(pk2bf(u0, u1), pk2bf(u2, u3));
        pr += 4 * N1;
      }
    } else {
#pragma unroll
      for (int jj = 0; jj < 8; ++jj) {
        int kb = k0 + jj * 4;
        float v0 = (kb + 0 < K) ? pr[0]        : 0.0f;
        float v1 = (kb + 1 < K) ? pr[N1]       : 0.0f;
        float v2 = (kb + 2 < K) ? pr[2 * N1]   : 0.0f;
        float v3 = (kb + 3 < K) ? pr[3 * N1]   : 0.0f;
        float u0 = (kb + 0 < K) ? pr[128]      : 0.0f;
        float u1 = (kb + 1 < K) ? pr[N1 + 128] : 0.0f;
        float u2 = (kb + 2 < K) ? pr[2 * N1 + 128] : 0.0f;
        float u3 = (kb + 3 < K) ? pr[3 * N1 + 128] : 0.0f;
        *(uint2*)&At[tid * 32 + jj * 4]         = make_uint2(pk2bf(v0, v1), pk2bf(v2, v3));
        *(uint2*)&At[(tid + 128) * 32 + jj * 4] = make_uint2(pk2bf(u0, u1), pk2bf(u2, u3));
        pr += 4 * N1;
      }
    }
    if (k0 + 32 < K)
      __builtin_prefetch(A + (long)(k0 + 32) * N + n0 + tid * 2, 0, 3);
    __syncthreads();

    // ---- fragments per CDNA5 ISA 7.12.2 ----
    union { v16bf v; uint4 q[2]; } fa0, fa1, fb;
    const unsigned short* wp0 = &Wt[row * 32 + half * 8];          // M rows 0..15
    fa0.q[0] = *(const uint4*)(wp0);
    fa0.q[1] = *(const uint4*)(wp0 + 16);
    const unsigned short* wp1 = &Wt[(16 + row) * 32 + half * 8];   // M rows 16..31
    fa1.q[0] = *(const uint4*)(wp1);
    fa1.q[1] = *(const uint4*)(wp1 + 16);

#pragma unroll
    for (int t = 0; t < 4; ++t) {
      const unsigned short* ap = &At[(wn + t * 16 + row) * 32 + half * 8];
      fb.q[0] = *(const uint4*)(ap);
      fb.q[1] = *(const uint4*)(ap + 16);
      acc[0][t] = __builtin_amdgcn_wmma_f32_16x16x32_bf16(false, fa0.v, false, fb.v,
                                                          (short)0, acc[0][t], false, false);
      acc[1][t] = __builtin_amdgcn_wmma_f32_16x16x32_bf16(false, fa1.v, false, fb.v,
                                                          (short)0, acc[1][t], false, false);
    }
    __syncthreads();
  }

  // D layout: lane (0-15|16-31) holds col n=lane&15, VGPR r -> row r (+8).
  const int col = lane & 15;
#pragma unroll
  for (int mt = 0; mt < 2; ++mt) {
#pragma unroll
    for (int t = 0; t < 4; ++t) {
#pragma unroll
      for (int r = 0; r < 8; ++r) {
        int m = m0 + mt * 16 + half * 8 + r;
        if (m < M) {
          float v = acc[mt][t][r];
          if (bias) v += bias[m];
          if (fuseGelu) v = gelu_exact(v);
          O[(long)m * N + n0 + wn + t * 16 + col] = v;
        }
      }
    }
  }
}

// =====================================================================
// Elementwise / data-movement kernels
// =====================================================================
__global__ void k_im2col2(const float* __restrict__ x, float* __restrict__ col,
                          long total, int C, int Zo, int Ho, int Wo)
{
  long i = (long)blockIdx.x * blockDim.x + threadIdx.x;
  if (i >= total) return;
  long So = (long)Zo * Ho * Wo;
  long s = i % So; long t = i / So;
  int ch = (int)(t % (C * 8)); int b = (int)(t / (C * 8));
  int c = ch >> 3, ijk = ch & 7;
  int di = (ijk >> 2) & 1, dj = (ijk >> 1) & 1, dk = ijk & 1;
  int zo = (int)(s / ((long)Ho * Wo)); long r = s % ((long)Ho * Wo);
  int ho = (int)(r / Wo), wo = (int)(r % Wo);
  int Zi = 2 * Zo, Hi = 2 * Ho, Wi = 2 * Wo;
  long xi = (((long)(b * C + c) * Zi + (2 * zo + di)) * Hi + (2 * ho + dj)) * Wi + (2 * wo + dk);
  col[i] = x[xi];
}

// g: (B, 8*Co, Sin) -> dst channels [0..Co) of (B, *, 8*Sin) with parity scatter (+bias)
__global__ void k_upscatter(const float* __restrict__ g, const float* __restrict__ bias,
                            float* __restrict__ dst, long dstBS, long total,
                            int Co, int Zi, int Hi, int Wi)
{
  long i = (long)blockIdx.x * blockDim.x + threadIdx.x;
  if (i >= total) return;
  long Sin = (long)Zi * Hi * Wi, So = 8 * Sin;
  long s2 = i % So; long t = i / So;
  int o = (int)(t % Co); int b = (int)(t / Co);
  int z2 = (int)(s2 / (4L * Hi * Wi)); long r = s2 % (4L * Hi * Wi);
  int h2 = (int)(r / (2 * Wi)), w2 = (int)(r % (2 * Wi));
  int zi = z2 >> 1, di = z2 & 1, hi = h2 >> 1, dj = h2 & 1, wi = w2 >> 1, dk = w2 & 1;
  long gidx = ((long)b * (8 * Co) + (di * 4 + dj * 2 + dk) * Co + o) * Sin
            + ((long)zi * Hi + hi) * Wi + wi;
  dst[(long)b * dstBS + (long)o * So + s2] = g[gidx] + bias[o];
}

__global__ void k_copy_chan(const float* __restrict__ src, float* __restrict__ dst,
                            long total, int Cs, long S, long srcBS, long dstBS, int chOff)
{
  long i = (long)blockIdx.x * blockDim.x + threadIdx.x;
  if (i >= total) return;
  long s = i % S; long t = i / S;
  int c = (int)(t % Cs); int b = (int)(t / Cs);
  dst[(long)b * dstBS + (long)(chOff + c) * S + s] = src[(long)b * srcBS + (long)c * S + s];
}

// uw (Cin, Co, 2,2,2) -> wt[(ijk*Co+o)*Cin + c]
__global__ void k_prep_upw(const float* __restrict__ uw, float* __restrict__ wt,
                           long total, int Cin, int Co)
{
  long i = (long)blockIdx.x * blockDim.x + threadIdx.x;
  if (i >= total) return;
  int c = (int)(i % Cin); long t = i / Cin;
  int o = (int)(t % Co); int ijk = (int)(t / Co);
  wt[i] = uw[((long)c * Co + o) * 8 + ijk];
}

// =====================================================================
// Truncated separable DFT pipeline (matches rfftn corner-mode truncation).
// One or two __sincosf per thread; inner loops use FMA phase-rotation recurrences.
// =====================================================================
__global__ void k_dft_w(const float* __restrict__ x, float* __restrict__ o,
                        long total, int W, int mw)
{
  long i = (long)blockIdx.x * blockDim.x + threadIdx.x;
  if (i >= total) return;
  int kw = (int)(i % mw); long row = i / mw;
  const float* p = x + row * W;
  float ss, cc; __sincosf(-6.28318530717958648f * (float)kw / (float)W, &ss, &cc);
  float cr = 1.f, ci = 0.f;          // e^{-i*2pi*kw*w/W}
  float sr = 0.f, si = 0.f;
  for (int w = 0; w < W; ++w) {
    float v = p[w];
    sr += v * cr; si += v * ci;
    float nc = cr * cc - ci * ss; ci = cr * ss + ci * cc; cr = nc;
  }
  o[2 * i] = sr; o[2 * i + 1] = si;
}

__global__ void k_dft_h(const float* __restrict__ in, float* __restrict__ o,
                        long total, int H, int mh, int mw)
{
  long i = (long)blockIdx.x * blockDim.x + threadIdx.x;
  if (i >= total) return;
  int kw = (int)(i % mw); long t = i / mw;
  int khi = (int)(t % (2 * mh)); long zrow = t / (2 * mh);
  int kh = (khi < mh) ? khi : (H - 2 * mh + khi);
  float ss, cc; __sincosf(-6.28318530717958648f * (float)kh / (float)H, &ss, &cc);
  float cr = 1.f, ci = 0.f;
  float sr = 0.f, si = 0.f;
  long b0 = (zrow * H) * (long)mw + kw;
  for (int h = 0; h < H; ++h) {
    long idx = b0 + (long)h * mw;
    float xr = in[2 * idx], xi = in[2 * idx + 1];
    sr += xr * cr - xi * ci;
    si += xr * ci + xi * cr;
    float nc = cr * cc - ci * ss; ci = cr * ss + ci * cc; cr = nc;
  }
  o[2 * i] = sr; o[2 * i + 1] = si;
}

__global__ void k_dft_z(const float* __restrict__ in, float* __restrict__ o,
                        long total, int Z, int mz, int mh, int mw)
{
  long i = (long)blockIdx.x * blockDim.x + threadIdx.x;
  if (i >= total) return;
  int kw = (int)(i % mw); long t = i / mw;
  int khi = (int)(t % (2 * mh)); long t2 = t / (2 * mh);
  int kzi = (int)(t2 % (2 * mz)); long bc = t2 / (2 * mz);
  int kz = (kzi < mz) ? kzi : (Z - 2 * mz + kzi);
  float ss, cc; __sincosf(-6.28318530717958648f * (float)kz / (float)Z, &ss, &cc);
  float cr = 1.f, ci = 0.f;
  float sr = 0.f, si = 0.f;
  long stride = (long)(2 * mh) * mw;
  long b0 = (bc * Z) * stride + (long)khi * mw + kw;
  for (int z = 0; z < Z; ++z) {
    long idx = b0 + (long)z * stride;
    float xr = in[2 * idx], xi = in[2 * idx + 1];
    sr += xr * cr - xi * ci;
    si += xr * ci + xi * cr;
    float nc = cr * cc - ci * ss; ci = cr * ss + ci * cc; cr = nc;
  }
  o[2 * i] = sr; o[2 * i + 1] = si;
}

// per-mode complex channel mix: out[b,o,kz,kh,kw] = sum_i x[b,i,..] * w[corner,i,o,..]
__global__ void k_specmix(const float* __restrict__ xf, const float* __restrict__ w,
                          float* __restrict__ o, long total,
                          int Ci, int Co, int mz, int mh, int mw)
{
  long i = (long)blockIdx.x * blockDim.x + threadIdx.x;
  if (i >= total) return;
  int kw = (int)(i % mw); long t = i / mw;
  int khi = (int)(t % (2 * mh)); long t2 = t / (2 * mh);
  int kzi = (int)(t2 % (2 * mz)); long t3 = t2 / (2 * mz);
  int oc = (int)(t3 % Co); int b = (int)(t3 / Co);
  int corner = ((kzi >= mz) ? 1 : 0) + ((khi >= mh) ? 2 : 0);
  int lz = (kzi >= mz) ? (kzi - mz) : kzi;
  int lh = (khi >= mh) ? (khi - mh) : khi;
  long xstep = (long)(2 * mz) * (2 * mh) * mw;
  long xidx = (((long)(b * Ci) * (2 * mz) + kzi) * (2 * mh) + khi) * mw + kw;
  long wstep = (long)Co * mz * mh * mw;
  long widx = ((((long)(corner * Ci) * Co + oc) * mz + lz) * mh + lh) * mw + kw;
  float orr = 0.f, oii = 0.f;
  for (int ci = 0; ci < Ci; ++ci) {
    float xr = xf[2 * xidx], xi = xf[2 * xidx + 1];
    float wr = w[2 * widx],  wi = w[2 * widx + 1];
    orr += xr * wr - xi * wi;
    oii += xr * wi + xi * wr;
    xidx += xstep; widx += wstep;
  }
  o[2 * i] = orr; o[2 * i + 1] = oii;
}

__global__ void k_idft_z(const float* __restrict__ in, float* __restrict__ o,
                         long total, int Z, int mz, int mh, int mw)
{
  long i = (long)blockIdx.x * blockDim.x + threadIdx.x;
  if (i >= total) return;
  int kw = (int)(i % mw); long t = i / mw;
  int khi = (int)(t % (2 * mh)); long t2 = t / (2 * mh);
  int z = (int)(t2 % Z); long bc = t2 / Z;
  long stride = (long)(2 * mh) * mw;
  long b0 = (bc * (2 * mz)) * stride + (long)khi * mw + kw;
  // per-step rotation e^{+i*2pi*z/Z}; positive block starts at phase 0,
  // negative block starts at phase 2pi*z*(Z-mz)/Z.
  float ss, cc; __sincosf(6.28318530717958648f * (float)z / (float)Z, &ss, &cc);
  float sr = 0.f, si = 0.f;
  float cr = 1.f, ci = 0.f;
  for (int kzi = 0; kzi < mz; ++kzi) {
    long idx = b0 + (long)kzi * stride;
    float ar = in[2 * idx], ai = in[2 * idx + 1];
    sr += ar * cr - ai * ci;
    si += ar * ci + ai * cr;
    float nc = cr * cc - ci * ss; ci = cr * ss + ci * cc; cr = nc;
  }
  __sincosf(6.28318530717958648f * (float)z * (float)(Z - mz) / (float)Z, &ci, &cr);
  for (int kzi = mz; kzi < 2 * mz; ++kzi) {
    long idx = b0 + (long)kzi * stride;
    float ar = in[2 * idx], ai = in[2 * idx + 1];
    sr += ar * cr - ai * ci;
    si += ar * ci + ai * cr;
    float nc = cr * cc - ci * ss; ci = cr * ss + ci * cc; cr = nc;
  }
  o[2 * i] = sr; o[2 * i + 1] = si;
}

__global__ void k_idft_h(const float* __restrict__ in, float* __restrict__ o,
                         long total, int H, int mh, int mw)
{
  long i = (long)blockIdx.x * blockDim.x + threadIdx.x;
  if (i >= total) return;
  int kw = (int)(i % mw); long t = i / mw;
  int h = (int)(t % H); long zrow = t / H;
  long b0 = (zrow * (2 * mh)) * (long)mw + kw;
  float ss, cc; __sincosf(6.28318530717958648f * (float)h / (float)H, &ss, &cc);
  float sr = 0.f, si = 0.f;
  float cr = 1.f, ci = 0.f;
  for (int khi = 0; khi < mh; ++khi) {
    long idx = b0 + (long)khi * mw;
    float ar = in[2 * idx], ai = in[2 * idx + 1];
    sr += ar * cr - ai * ci;
    si += ar * ci + ai * cr;
    float nc = cr * cc - ci * ss; ci = cr * ss + ci * cc; cr = nc;
  }
  __sincosf(6.28318530717958648f * (float)h * (float)(H - mh) / (float)H, &ci, &cr);
  for (int khi = mh; khi < 2 * mh; ++khi) {
    long idx = b0 + (long)khi * mw;
    float ar = in[2 * idx], ai = in[2 * idx + 1];
    sr += ar * cr - ai * ci;
    si += ar * ci + ai * cr;
    float nc = cr * cc - ci * ss; ci = cr * ss + ci * cc; cr = nc;
  }
  o[2 * i] = sr; o[2 * i + 1] = si;
}

// inverse rfft along W with Hermitian factor-2, full 1/(Z*H*W) normalization
__global__ void k_idft_w(const float* __restrict__ in, float* __restrict__ o,
                         long total, int W, int mw, float invN)
{
  long i = (long)blockIdx.x * blockDim.x + threadIdx.x;
  if (i >= total) return;
  int w = (int)(i % W); long row = i / W;
  long b0 = row * mw;
  float ss, cc; __sincosf(6.28318530717958648f * (float)w / (float)W, &ss, &cc);
  float acc = in[2 * b0];            // kw = 0 term (factor 1)
  float cr = cc, ci = ss;            // phase for kw = 1
  for (int kw = 1; kw < mw; ++kw) {
    float ar = in[2 * (b0 + kw)], ai = in[2 * (b0 + kw) + 1];
    acc += 2.0f * (ar * cr - ai * ci);
    float nc = cr * cc - ci * ss; ci = cr * ss + ci * cc; cr = nc;
  }
  o[i] = acc * invN;
}

// fused: v = a+c; InstanceNorm over S (biased var, eps 1e-5); GELU. One block per (b,ch).
__global__ __launch_bounds__(256)
void k_add_inorm_gelu(const float* __restrict__ a, const float* __restrict__ c,
                      float* __restrict__ o, int S)
{
  __shared__ float r1[256], r2[256];
  long base = (long)blockIdx.x * S;
  int tid = threadIdx.x;
  float sum = 0.f, sq = 0.f;
  for (int i = tid; i < S; i += 256) {
    float v = a[base + i] + c[base + i];
    sum += v; sq += v * v;
  }
  r1[tid] = sum; r2[tid] = sq;
  __syncthreads();
  for (int st = 128; st > 0; st >>= 1) {
    if (tid < st) { r1[tid] += r1[tid + st]; r2[tid] += r2[tid + st]; }
    __syncthreads();
  }
  float mean = r1[0] / (float)S;
  float var  = r2[0] / (float)S - mean * mean;
  float inv  = rsqrtf(var + 1e-5f);
  for (int i = tid; i < S; i += 256) {
    float v = (a[base + i] + c[base + i] - mean) * inv;
    o[base + i] = gelu_exact(v);
  }
}

// =====================================================================
// Host-side orchestration
// =====================================================================
struct Arena {
  char* base; size_t off; size_t cap;
  void* alloc(size_t bytes) {
    bytes = (bytes + 255) & ~(size_t)255;
    void* p = base + off; off += bytes; return p;
  }
  float* af(long n) { return (float*)alloc(sizeof(float) * (size_t)n); }
};

static inline dim3 gs(long total) { return dim3((unsigned)((total + TPB - 1) / TPB)); }

static void gemm(hipStream_t st, const float* act, long aBS, const float* w,
                 const float* bias, float* out, long oBS,
                 int B, int M, int K, int N, bool gelu)
{
  int Kpad = (K + 31) & ~31;
  dim3 grid((unsigned)(N / 256), (unsigned)((M + 31) / 32), (unsigned)B);
  k_gemm_wmma<<<grid, 128, 0, st>>>(act, aBS, w, bias, out, oBS, M, K, Kpad, N, gelu ? 1 : 0);
}

// FNO block: out = GELU(InstanceNorm(spec(x) + conv1x1(x)))
static void launch_fno(hipStream_t st, Arena& ar, const float* x, float* out,
                       int B, int Ci, int Co, int Z, int H, int W,
                       int mz, int mh, int mw,
                       const float* sw, const float* cw, const float* cb)
{
  size_t mark = ar.off;
  long S  = (long)Z * H * W;
  long BC = (long)B * Ci;
  long ZH = (long)Z * H;

  float* t1 = ar.af(BC * ZH * mw * 2);
  { long tot = BC * ZH * mw;
    k_dft_w<<<gs(tot), TPB, 0, st>>>(x, t1, tot, W, mw); }

  float* t2 = ar.af(BC * Z * (2L * mh) * mw * 2);
  { long tot = BC * Z * 2L * mh * mw;
    k_dft_h<<<gs(tot), TPB, 0, st>>>(t1, t2, tot, H, mh, mw); }

  float* t3 = ar.af(BC * (2L * mz) * (2L * mh) * mw * 2);
  { long tot = BC * 2L * mz * 2L * mh * mw;
    k_dft_z<<<gs(tot), TPB, 0, st>>>(t2, t3, tot, Z, mz, mh, mw); }

  float* t4 = ar.af((long)B * Co * (2L * mz) * (2L * mh) * mw * 2);
  { long tot = (long)B * Co * 2L * mz * 2L * mh * mw;
    k_specmix<<<gs(tot), TPB, 0, st>>>(t3, sw, t4, tot, Ci, Co, mz, mh, mw); }

  float* t5 = ar.af((long)B * Co * Z * (2L * mh) * mw * 2);
  { long tot = (long)B * Co * Z * 2L * mh * mw;
    k_idft_z<<<gs(tot), TPB, 0, st>>>(t4, t5, tot, Z, mz, mh, mw); }

  float* t6 = ar.af((long)B * Co * ZH * mw * 2);
  { long tot = (long)B * Co * ZH * mw;
    k_idft_h<<<gs(tot), TPB, 0, st>>>(t5, t6, tot, H, mh, mw); }

  float* spec = ar.af((long)B * Co * S);
  { long tot = (long)B * Co * S;
    k_idft_w<<<gs(tot), TPB, 0, st>>>(t6, spec, tot, W, mw, 1.0f / (float)S); }

  float* conv = ar.af((long)B * Co * S);
  gemm(st, x, (long)Ci * S, cw, cb, conv, (long)Co * S, B, Co, Ci, (int)S, false);

  k_add_inorm_gelu<<<dim3((unsigned)(B * Co)), 256, 0, st>>>(spec, conv, out, (int)S);
  ar.off = mark;
}

static void down2(hipStream_t st, Arena& ar, const float* x, float* out,
                  int B, int C, int Zo, const float* dw, const float* db)
{
  size_t mark = ar.off;
  long So = (long)Zo * Zo * Zo;
  float* col = ar.af((long)B * C * 8 * So);
  long tot = (long)B * C * 8 * So;
  k_im2col2<<<gs(tot), TPB, 0, st>>>(x, col, tot, C, Zo, Zo, Zo);
  gemm(st, col, (long)C * 8 * So, dw, db, out, (long)C * So, B, C, C * 8, (int)So, false);
  ar.off = mark;
}

// up2 into channels [0..Co) of dst (dst batch stride given); adds bias.
static void up2(hipStream_t st, Arena& ar, const float* x, float* dst, long dstBS,
                int B, int Cin, int Co, int Zi, const float* uw, const float* ub)
{
  size_t mark = ar.off;
  long Sin = (long)Zi * Zi * Zi;
  float* wT = ar.af((long)Cin * Co * 8);
  { long tot = (long)Cin * Co * 8;
    k_prep_upw<<<gs(tot), TPB, 0, st>>>(uw, wT, tot, Cin, Co); }
  float* g = ar.af((long)B * 8 * Co * Sin);
  gemm(st, x, (long)Cin * Sin, wT, nullptr, g, (long)8 * Co * Sin, B, 8 * Co, Cin, (int)Sin, false);
  { long tot = (long)B * Co * 8 * Sin;
    k_upscatter<<<gs(tot), TPB, 0, st>>>(g, ub, dst, dstBS, tot, Co, Zi, Zi, Zi); }
  ar.off = mark;
}

extern "C" void kernel_launch(void* const* d_in, const int* in_sizes, int n_in,
                              void* d_out, int out_size, void* d_ws, size_t ws_size,
                              hipStream_t stream)
{
  (void)in_sizes; (void)n_in; (void)out_size;
  const float* x      = (const float*)d_in[0];
  const float* lift_w = (const float*)d_in[1];
  const float* lift_b = (const float*)d_in[2];
  const float* e1_sw = (const float*)d_in[3],  *e1_cw = (const float*)d_in[4],
             * e1_cb = (const float*)d_in[5],  *e1_dw = (const float*)d_in[6],
             * e1_db = (const float*)d_in[7];
  const float* e2_sw = (const float*)d_in[8],  *e2_cw = (const float*)d_in[9],
             * e2_cb = (const float*)d_in[10], *e2_dw = (const float*)d_in[11],
             * e2_db = (const float*)d_in[12];
  const float* e3_sw = (const float*)d_in[13], *e3_cw = (const float*)d_in[14],
             * e3_cb = (const float*)d_in[15], *e3_dw = (const float*)d_in[16],
             * e3_db = (const float*)d_in[17];
  const float* bn_sw = (const float*)d_in[18], *bn_cw = (const float*)d_in[19],
             * bn_cb = (const float*)d_in[20];
  const float* d3_uw = (const float*)d_in[21], *d3_ub = (const float*)d_in[22],
             * d3_sw = (const float*)d_in[23], *d3_cw = (const float*)d_in[24],
             * d3_cb = (const float*)d_in[25];
  const float* d2_uw = (const float*)d_in[26], *d2_ub = (const float*)d_in[27],
             * d2_sw = (const float*)d_in[28], *d2_cw = (const float*)d_in[29],
             * d2_cb = (const float*)d_in[30];
  const float* d1_uw = (const float*)d_in[31], *d1_ub = (const float*)d_in[32],
             * d1_sw = (const float*)d_in[33], *d1_cw = (const float*)d_in[34],
             * d1_cb = (const float*)d_in[35];
  const float* p1_w = (const float*)d_in[36], *p1_b = (const float*)d_in[37];
  const float* p2_w = (const float*)d_in[38], *p2_b = (const float*)d_in[39];

  Arena ar{(char*)d_ws, 0, ws_size};

  const int B = 2;
  const long S1 = 64L * 64 * 64;   // 64^3
  const long S2 = 32L * 32 * 32;   // 32^3
  const long S3 = 16L * 16 * 16;   // 16^3
  const long S4 = 8L * 8 * 8;      // 8^3

  // ---- lift: (B,20,S1) -> (B,32,S1)
  float* h1 = ar.af((long)B * 32 * S1);
  gemm(stream, x, 20L * S1, lift_w, lift_b, h1, 32L * S1, B, 32, 20, (int)S1, false);

  // ---- encoder 1
  float* s1 = ar.af((long)B * 32 * S1);
  launch_fno(stream, ar, h1, s1, B, 32, 32, 64, 64, 64, 12, 12, 12, e1_sw, e1_cw, e1_cb);
  float* h2 = ar.af((long)B * 32 * S2);
  down2(stream, ar, s1, h2, B, 32, 32, e1_dw, e1_db);

  // ---- encoder 2
  float* s2 = ar.af((long)B * 64 * S2);
  launch_fno(stream, ar, h2, s2, B, 32, 64, 32, 32, 32, 6, 6, 6, e2_sw, e2_cw, e2_cb);
  float* h3 = ar.af((long)B * 64 * S3);
  down2(stream, ar, s2, h3, B, 64, 16, e2_dw, e2_db);

  // ---- encoder 3
  float* s3 = ar.af((long)B * 128 * S3);
  launch_fno(stream, ar, h3, s3, B, 64, 128, 16, 16, 16, 3, 3, 3, e3_sw, e3_cw, e3_cb);
  float* h4 = ar.af((long)B * 128 * S4);
  down2(stream, ar, s3, h4, B, 128, 8, e3_dw, e3_db);

  // ---- bottleneck
  float* hb = ar.af((long)B * 128 * S4);
  launch_fno(stream, ar, h4, hb, B, 128, 128, 8, 8, 8, 3, 3, 3, bn_sw, bn_cw, bn_cb);

  // ---- decoder 3: up(128->64, 8->16), concat s3 (128) => 192ch
  float* cat3 = ar.af((long)B * 192 * S3);
  up2(stream, ar, hb, cat3, 192L * S3, B, 128, 64, 8, d3_uw, d3_ub);
  { long tot = (long)B * 128 * S3;
    k_copy_chan<<<gs(tot), TPB, 0, stream>>>(s3, cat3, tot, 128, S3, 128L * S3, 192L * S3, 64); }
  float* d3o = ar.af((long)B * 64 * S3);
  launch_fno(stream, ar, cat3, d3o, B, 192, 64, 16, 16, 16, 3, 3, 3, d3_sw, d3_cw, d3_cb);

  // ---- decoder 2: up(64->32, 16->32), concat s2 (64) => 96ch
  float* cat2 = ar.af((long)B * 96 * S2);
  up2(stream, ar, d3o, cat2, 96L * S2, B, 64, 32, 16, d2_uw, d2_ub);
  { long tot = (long)B * 64 * S2;
    k_copy_chan<<<gs(tot), TPB, 0, stream>>>(s2, cat2, tot, 64, S2, 64L * S2, 96L * S2, 32); }
  float* d2o = ar.af((long)B * 32 * S2);
  launch_fno(stream, ar, cat2, d2o, B, 96, 32, 32, 32, 32, 6, 6, 6, d2_sw, d2_cw, d2_cb);

  // ---- decoder 1: up(32->32, 32->64), concat s1 (32) => 64ch
  float* cat1 = ar.af((long)B * 64 * S1);
  up2(stream, ar, d2o, cat1, 64L * S1, B, 32, 32, 32, d1_uw, d1_ub);
  { long tot = (long)B * 32 * S1;
    k_copy_chan<<<gs(tot), TPB, 0, stream>>>(s1, cat1, tot, 32, S1, 32L * S1, 64L * S1, 32); }
  float* d1o = ar.af((long)B * 32 * S1);
  launch_fno(stream, ar, cat1, d1o, B, 64, 32, 64, 64, 64, 12, 12, 12, d1_sw, d1_cw, d1_cb);

  // ---- projection: GELU(c1x1(.,p1)) then c1x1(.,p2) -> d_out (B,20,S1)
  float* p1o = ar.af((long)B * 64 * S1);
  gemm(stream, d1o, 32L * S1, p1_w, p1_b, p1o, 64L * S1, B, 64, 32, (int)S1, true);
  gemm(stream, p1o, 64L * S1, p2_w, p2_b, (float*)d_out, 20L * S1, B, 20, 64, (int)S1, false);
}
